// FAEncoder_68745246540384
// MI455X (gfx1250) — compile-verified, matrix operand order checked by reading, not run on hardware
//
#include <hip/hip_runtime.h>
#include <hip/hip_bf16.h>
#include <math.h>

typedef __attribute__((ext_vector_type(16))) __bf16 v16bf;
typedef __attribute__((ext_vector_type(8)))  __bf16 v8bf;
typedef __attribute__((ext_vector_type(8)))  float  v8f;

#define LSEQ 512
#define BCOLS 64           // B*8 symmetry ops
#define D2F 128            // feature dim per layer (D2)
#define HDIM 256
#define NROWS (LSEQ * BCOLS)   // 32768
#define K0PAD 288          // 259 padded to multiple of 32

// ---------------------------------------------------------------------------
// WMMA helpers (CDNA5 wave32 fragment layouts, ISA 7.12.2)
// element e of a 16-bit A(16xK)/B(Kx16) fragment -> K offset in a 32-chunk
// ---------------------------------------------------------------------------
__device__ __forceinline__ int fragK(int e, int laneHi) {
    return ((e >> 3) << 4) + (laneHi << 3) + (e & 7);
}

__device__ __forceinline__ v16bf combine16(v8bf lo, v8bf hi) {
    return __builtin_shufflevector(lo, hi, 0,1,2,3,4,5,6,7,8,9,10,11,12,13,14,15);
}

// load a row-major bf16 A/B^T fragment: base points at (row, kc + 8*laneHi)
__device__ __forceinline__ v16bf load_frag_rm(const __bf16* base) {
    v8bf lo = *(const v8bf*)(base);
    v8bf hi = *(const v8bf*)(base + 16);
    return combine16(lo, hi);
}

__device__ __forceinline__ v8f wmma_bf16(v16bf a, v16bf b, v8f c) {
    return __builtin_amdgcn_wmma_f32_16x16x32_bf16(
        false, a, false, b, (short)0, c, false, false);
}

__device__ __forceinline__ float grp_max16(float v) {
#pragma unroll
    for (int m = 1; m <= 8; m <<= 1) v = fmaxf(v, __shfl_xor(v, m, 32));
    return v;
}
__device__ __forceinline__ float grp_sum16(float v) {
#pragma unroll
    for (int m = 1; m <= 8; m <<= 1) v += __shfl_xor(v, m, 32);
    return v;
}

// ---------------------------------------------------------------------------
// Stage 0a: masked center + 3x3 covariance + analytic eigh, one thread per b
// ---------------------------------------------------------------------------
__device__ void eigvec3(float a00, float a01, float a02, float a11, float a12,
                        float a22, float lam, float out[3]) {
    float r0[3] = {a00 - lam, a01, a02};
    float r1[3] = {a01, a11 - lam, a12};
    float r2[3] = {a02, a12, a22 - lam};
    float c0[3] = {r0[1]*r1[2]-r0[2]*r1[1], r0[2]*r1[0]-r0[0]*r1[2], r0[0]*r1[1]-r0[1]*r1[0]};
    float c1[3] = {r0[1]*r2[2]-r0[2]*r2[1], r0[2]*r2[0]-r0[0]*r2[2], r0[0]*r2[1]-r0[1]*r2[0]};
    float c2[3] = {r1[1]*r2[2]-r1[2]*r2[1], r1[2]*r2[0]-r1[0]*r2[2], r1[0]*r2[1]-r1[1]*r2[0]};
    float n0 = c0[0]*c0[0]+c0[1]*c0[1]+c0[2]*c0[2];
    float n1 = c1[0]*c1[0]+c1[1]*c1[1]+c1[2]*c1[2];
    float n2 = c2[0]*c2[0]+c2[1]*c2[1]+c2[2]*c2[2];
    const float* best = c0; float nb = n0;
    if (n1 > nb) { best = c1; nb = n1; }
    if (n2 > nb) { best = c2; nb = n2; }
    float inv = rsqrtf(fmaxf(nb, 1e-30f));
    out[0] = best[0]*inv; out[1] = best[1]*inv; out[2] = best[2]*inv;
}

__device__ void signfix3(float v[3]) {
    float a0 = fabsf(v[0]), a1 = fabsf(v[1]), a2 = fabsf(v[2]);
    float m = (a0 >= a1 && a0 >= a2) ? v[0] : (a1 >= a2 ? v[1] : v[2]);
    if (m < 0.f) { v[0] = -v[0]; v[1] = -v[1]; v[2] = -v[2]; }
}

__global__ void eigh_kernel(const float* __restrict__ cords,
                            const float* __restrict__ mask,
                            float* __restrict__ Vout,   // [8][3][3]
                            float* __restrict__ cenOut) // [8][3]
{
    int b = threadIdx.x;
    if (b >= 8) return;
    float cx = 0.f, cy = 0.f, cz = 0.f, sm = 0.f;
    for (int n = 0; n < LSEQ; ++n) {
        float mk = mask[b*LSEQ + n];
        const float* X = cords + ((size_t)(b*LSEQ + n)*4 + 1)*3;
        cx += mk*X[0]; cy += mk*X[1]; cz += mk*X[2]; sm += mk;
    }
    cx /= sm; cy /= sm; cz /= sm;
    float a00=0,a01=0,a02=0,a11=0,a12=0,a22=0;
    for (int n = 0; n < LSEQ; ++n) {
        float mk = mask[b*LSEQ + n];
        const float* X = cords + ((size_t)(b*LSEQ + n)*4 + 1)*3;
        float x = X[0]-cx*mk, y = X[1]-cy*mk, z = X[2]-cz*mk;
        a00 += x*x; a01 += x*y; a02 += x*z; a11 += y*y; a12 += y*z; a22 += z*z;
    }
    float q  = (a00+a11+a22)/3.f;
    float p1 = a01*a01 + a02*a02 + a12*a12;
    float b00 = a00-q, b11 = a11-q, b22 = a22-q;
    float p2 = b00*b00 + b11*b11 + b22*b22 + 2.f*p1;
    float p  = sqrtf(p2/6.f);
    float V[3][3];
    if (p < 1e-12f) {
        V[0][0]=1;V[0][1]=0;V[0][2]=0; V[1][0]=0;V[1][1]=1;V[1][2]=0; V[2][0]=0;V[2][1]=0;V[2][2]=1;
    } else {
        float inv = 1.f/p;
        float c00=b00*inv, c01=a01*inv, c02=a02*inv, c11=b11*inv, c12=a12*inv, c22=b22*inv;
        float detB = c00*(c11*c22 - c12*c12) - c01*(c01*c22 - c12*c02) + c02*(c01*c12 - c11*c02);
        float r = fminf(1.f, fmaxf(-1.f, 0.5f*detB));
        float phi = acosf(r)/3.f;
        float eLg = q + 2.f*p*cosf(phi);
        float eSm = q + 2.f*p*cosf(phi + 2.0943951023931953f);
        float vL[3], vS[3], vM[3];
        eigvec3(a00,a01,a02,a11,a12,a22, eLg, vL);
        eigvec3(a00,a01,a02,a11,a12,a22, eSm, vS);
        vM[0] = vL[1]*vS[2]-vL[2]*vS[1];
        vM[1] = vL[2]*vS[0]-vL[0]*vS[2];
        vM[2] = vL[0]*vS[1]-vL[1]*vS[0];
        signfix3(vL); signfix3(vS); signfix3(vM);
        for (int j = 0; j < 3; ++j) { V[j][0] = vS[j]; V[j][1] = vM[j]; V[j][2] = vL[j]; }
    }
    for (int j = 0; j < 3; ++j)
        for (int i = 0; i < 3; ++i) Vout[b*9 + j*3 + i] = V[j][i];
    cenOut[b*3+0] = cx; cenOut[b*3+1] = cy; cenOut[b*3+2] = cz;
}

// ---------------------------------------------------------------------------
// Stage 0b: key-pad flags [8][512]
// ---------------------------------------------------------------------------
__global__ void pad_kernel(const float* __restrict__ mask, float* __restrict__ padflag) {
    int i = blockIdx.x*blockDim.x + threadIdx.x;
    if (i < 8*LSEQ) padflag[i] = (mask[i] == 0.f) ? 1.f : 0.f;
}

// ---------------------------------------------------------------------------
// Stage 0c: build h0 directly in bf16 (L, 64, 288)
// ---------------------------------------------------------------------------
__global__ void build_h0(const int* __restrict__ seqs, const float* __restrict__ cords,
                         const float* __restrict__ mask, const float* __restrict__ emb,
                         const float* __restrict__ Vmat, const float* __restrict__ cen,
                         __bf16* __restrict__ hbufb)
{
    size_t i = (size_t)blockIdx.x*blockDim.x + threadIdx.x;
    const size_t total = (size_t)NROWS * K0PAD;
    if (i >= total) return;
    int    c   = (int)(i % K0PAD);
    size_t row = i / K0PAD;
    int bcol = (int)(row & 63);
    int l    = (int)(row >> 6);
    int bb = bcol >> 3, op = bcol & 7;
    float val = 0.f;
    if (c < 3) {
        float mk = mask[bb*LSEQ + l];
        const float* X  = cords + ((size_t)(bb*LSEQ + l)*4 + 1)*3;
        const float* V  = Vmat + bb*9;
        const float* ce = cen  + bb*3;
        float acc = 0.f;
#pragma unroll
        for (int j = 0; j < 3; ++j) {
            float sgn = ((op >> (2 - j)) & 1) ? 1.f : -1.f;
            acc += sgn * V[j*3 + c] * (X[j] - ce[j]*mk);
        }
        val = acc;
    } else if (c < 259) {
        int s = seqs[bb*LSEQ + l];
        if (s < 0) s = 82;
        val = emb[(size_t)s*256 + (c - 3)];
    }
    hbufb[row*K0PAD + c] = (__bf16)val;
}

// ---------------------------------------------------------------------------
// One-time weight pack into fragment order: Wpk[kc][ntile][lane][e] (zero-padded)
// ---------------------------------------------------------------------------
__global__ void pack_weight(const float* __restrict__ W, __bf16* __restrict__ Wpk,
                            int Kact, int Kchunks, int Ntiles, int N)
{
    int i = blockIdx.x*blockDim.x + threadIdx.x;
    int total = Kchunks*Ntiles*512;
    if (i >= total) return;
    int e    = i & 15;
    int lane = (i >> 4) & 31;
    int rest = i >> 9;
    int nt   = rest % Ntiles;
    int kc   = rest / Ntiles;
    int laneHi = lane >> 4;
    int k   = kc*32 + fragK(e, laneHi);
    int col = nt*16 + (lane & 15);
    float v = (k < Kact) ? W[(size_t)k*N + col] : 0.f;
    Wpk[i] = (__bf16)v;
}

// ---------------------------------------------------------------------------
// One-time V pack into B-fragment order: vpk[b][kc16][nt8][lane][e]
// ---------------------------------------------------------------------------
__global__ void pack_v(const float* __restrict__ vb, __bf16* __restrict__ vpk) {
    int i = blockIdx.x*blockDim.x + threadIdx.x;
    if (i >= BCOLS*16*8*512) return;
    int e    = i & 15;
    int lane = (i >> 4) & 31;
    int nt   = (i >> 9) & 7;
    int kc   = (i >> 12) & 15;
    int b    = i >> 16;
    int laneHi = lane >> 4;
    int krow = kc*32 + fragK(e, laneHi);
    int col  = nt*16 + (lane & 15);
    vpk[i] = (__bf16)vb[((size_t)krow*BCOLS + b)*D2F + col];
}

// ---------------------------------------------------------------------------
// WMMA GEMM: C[M x N] = A[M x K] * B  (A: bf16 row-major, B: packed fragments)
// One wave: 16x64 output tile. All 4 B-fragments of a k-chunk are loaded
// before the first WMMA consumes any, so loads overlap the WMMA pipe.
// ---------------------------------------------------------------------------
__global__ __launch_bounds__(32)
void gemm_bf16(const __bf16* __restrict__ A, int lda, int Kpad,
               const __bf16* __restrict__ Bpk, int Ntiles,
               float* __restrict__ C, int ldc, __bf16* __restrict__ Cb)
{
    const int lane   = threadIdx.x & 31;
    const int laneHi = lane >> 4;
    const int lanelo = lane & 15;
    const int m0 = blockIdx.x * 16;
    const int ntBase = blockIdx.y * 4;
    v8f acc[4] = {};
    const __bf16* arow = A + (size_t)(m0 + lanelo)*lda + (laneHi << 3);
    for (int kc = 0; kc < Kpad; kc += 32) {
        __builtin_prefetch(arow + kc + 128, 0, 1);
        v16bf a = load_frag_rm(arow + kc);
        const __bf16* bp = Bpk + (((size_t)(kc >> 5)*Ntiles + ntBase)*32 + lane)*16;
        v16bf bfr[4];
#pragma unroll
        for (int t = 0; t < 4; ++t) bfr[t] = *(const v16bf*)(bp + (size_t)t*512);
#pragma unroll
        for (int t = 0; t < 4; ++t) acc[t] = wmma_bf16(a, bfr[t], acc[t]);
    }
#pragma unroll
    for (int t = 0; t < 4; ++t) {
        int col = (ntBase + t)*16 + lanelo;
#pragma unroll
        for (int r = 0; r < 8; ++r) {
            int row = m0 + r + (laneHi << 3);
            float v = acc[t][r];
            C[(size_t)row*ldc + col] = v;
            if (Cb) Cb[(size_t)row*ldc + col] = (__bf16)v;
        }
    }
}

// ---------------------------------------------------------------------------
// Flash-style attention: a[l,b,:] = softmax_m(q.k^T/sqrt(128), masked) @ v + q
// One wave: 16 query rows x 128 features, streaming keys in 32-chunks.
// All k-fragments of a chunk are loaded before the score WMMAs; the V-operand
// fragments are issued before the softmax/LDS work so they are in flight
// during the exp/reduction VALU phase.
// ---------------------------------------------------------------------------
__global__ __launch_bounds__(32)
void attn_kernel(const __bf16* __restrict__ qbb, const __bf16* __restrict__ kbb,
                 const __bf16* __restrict__ vpk, const float* __restrict__ qf32,
                 const float* __restrict__ padflag,
                 float* __restrict__ ab, __bf16* __restrict__ abb)
{
    alignas(32) __shared__ __bf16 Plds[16 * 32];
    const int lane   = threadIdx.x & 31;
    const int laneHi = lane >> 4;
    const int lanelo = lane & 15;
    const int l0 = blockIdx.x * 16;
    const int b  = blockIdx.y;           // 0..63
    const int bb = b >> 3;
    const float scale = 0.08838834764831845f;  // 1/sqrt(128)

    // preload q A-fragments (4 chunks of K=32)
    v16bf qf[4];
    const __bf16* qrow = qbb + ((size_t)(l0 + lanelo)*BCOLS + b)*D2F + (laneHi << 3);
#pragma unroll
    for (int kc = 0; kc < 4; ++kc) qf[kc] = load_frag_rm(qrow + kc*32);

    v8f O[8] = {};
    float m_s[8], l_s[8];
#pragma unroll
    for (int r = 0; r < 8; ++r) { m_s[r] = -3.0e38f; l_s[r] = 0.f; }

    for (int m0 = 0; m0 < LSEQ; m0 += 32) {
        // ---- load all k fragments for this chunk, then 8 score WMMAs
        const __bf16* krow0 = kbb + ((size_t)(m0 + lanelo)*BCOLS + b)*D2F + (laneHi << 3);
        const __bf16* krow1 = krow0 + (size_t)16*BCOLS*D2F;
        v16bf kf0[4], kf1[4];
#pragma unroll
        for (int kc = 0; kc < 4; ++kc) {
            kf0[kc] = load_frag_rm(krow0 + kc*32);
            kf1[kc] = load_frag_rm(krow1 + kc*32);
        }
        v8f S0 = {}, S1 = {};
#pragma unroll
        for (int kc = 0; kc < 4; ++kc) {
            S0 = wmma_bf16(qf[kc], kf0[kc], S0);
            S1 = wmma_bf16(qf[kc], kf1[kc], S1);
        }
        // ---- issue V-operand fragment loads early (independent of softmax)
        const __bf16* vbase = vpk + ((((size_t)b*16 + (m0 >> 5))*8)*32 + lane)*16;
        v16bf vfr[8];
#pragma unroll
        for (int t = 0; t < 8; ++t) vfr[t] = *(const v16bf*)(vbase + (size_t)t*512);
        // ---- key-padding mask + online softmax (row stats per 16-lane group)
        float pad0 = padflag[bb*LSEQ + m0 + lanelo];
        float pad1 = padflag[bb*LSEQ + m0 + 16 + lanelo];
        float p0a[8], p1a[8];
#pragma unroll
        for (int r = 0; r < 8; ++r) {
            float s0 = (pad0 != 0.f) ? -1.0e9f : S0[r]*scale;
            float s1 = (pad1 != 0.f) ? -1.0e9f : S1[r]*scale;
            float rm = grp_max16(fmaxf(s0, s1));
            float mn = fmaxf(m_s[r], rm);
            float alpha = __expf(m_s[r] - mn);
            float e0 = __expf(s0 - mn);
            float e1 = __expf(s1 - mn);
            float rs = grp_sum16(e0 + e1);
            l_s[r] = l_s[r]*alpha + rs;
            m_s[r] = mn;
            p0a[r] = e0; p1a[r] = e1;
#pragma unroll
            for (int t = 0; t < 8; ++t) O[t][r] *= alpha;
        }
        // ---- C-layout P tile -> LDS -> A-fragment
        __syncthreads();
#pragma unroll
        for (int r = 0; r < 8; ++r) {
            int row = r + (laneHi << 3);
            Plds[row*32 + lanelo]      = (__bf16)p0a[r];
            Plds[row*32 + 16 + lanelo] = (__bf16)p1a[r];
        }
        __syncthreads();
        v8bf plo = *(const v8bf*)&Plds[lanelo*32 + (laneHi << 3)];
        v8bf phi = *(const v8bf*)&Plds[lanelo*32 + 16 + (laneHi << 3)];
        v16bf pa = combine16(plo, phi);
        // ---- O += P @ V  (8 feature tiles)
#pragma unroll
        for (int t = 0; t < 8; ++t) O[t] = wmma_bf16(pa, vfr[t], O[t]);
    }
    // normalize + residual q; write f32 and bf16 copies
#pragma unroll
    for (int t = 0; t < 8; ++t)
#pragma unroll
        for (int r = 0; r < 8; ++r) {
            int row = l0 + r + (laneHi << 3);
            size_t idx = ((size_t)row*BCOLS + b)*D2F + t*16 + lanelo;
            float inv = (l_s[r] > 0.f) ? 1.f/l_s[r] : 0.f;
            float v = O[t][r]*inv + qf32[idx];
            ab[idx]  = v;
            abb[idx] = (__bf16)v;
        }
}

// ---------------------------------------------------------------------------
// Bidirectional SRU scan: one thread per (batch-col, dir, channel)
// U layout (L,64,2,4,128); writes h2 f32 (L,64,256) + bf16 copy
// ---------------------------------------------------------------------------
__global__ void sru_scan(const float* __restrict__ U, const float* __restrict__ padflag,
                         const float* __restrict__ vf, const float* __restrict__ vr,
                         const float* __restrict__ bf, const float* __restrict__ br,
                         float* __restrict__ h2, __bf16* __restrict__ h2b)
{
    int t = blockIdx.x*blockDim.x + threadIdx.x;
    if (t >= BCOLS*2*D2F) return;
    int d   = t & 127;
    int dir = (t >> 7) & 1;
    int bt  = t >> 8;
    int bb  = bt >> 3;
    float vfd = vf[dir*D2F + d], vrd = vr[dir*D2F + d];
    float bfd = bf[dir*D2F + d], brd = br[dir*D2F + d];
    float c = 0.f;
    for (int step = 0; step < LSEQ; ++step) {
        int l = dir ? (LSEQ - 1 - step) : step;
        size_t base = (((size_t)l*BCOLS + bt)*8 + dir*4)*D2F + d;
        float u0 = U[base];
        float u1 = U[base + D2F];
        float u2 = U[base + 2*D2F];
        float u3 = U[base + 3*D2F];
        float p  = padflag[bb*LSEQ + l];
        float f  = 1.f/(1.f + __expf(-(u1 + vfd*c + bfd)));
        float r  = 1.f/(1.f + __expf(-(u2 + vrd*c + brd)));
        float cn = f*c + (1.f - f)*u0;
        cn = (p != 0.f) ? c : cn;
        float h = (p != 0.f) ? 0.f : (r*cn + (1.f - r)*u3);
        c = cn;
        size_t oidx = ((size_t)l*BCOLS + bt)*HDIM + dir*D2F + d;
        h2[oidx]  = h;
        h2b[oidx] = (__bf16)h;
    }
}

// ---------------------------------------------------------------------------
// Mean over the 8 symmetry ops -> (B, N, 256)
// ---------------------------------------------------------------------------
__global__ void finalize_mean(const float* __restrict__ h2, float* __restrict__ out) {
    int i = blockIdx.x*blockDim.x + threadIdx.x;
    if (i >= 8*LSEQ*HDIM) return;
    int c  = i & 255;
    int n  = (i >> 8) & 511;
    int bb = i >> 17;
    float s = 0.f;
#pragma unroll
    for (int op = 0; op < 8; ++op)
        s += h2[((size_t)n*BCOLS + bb*8 + op)*HDIM + c];
    out[i] = s * 0.125f;
}

// ---------------------------------------------------------------------------
// Host orchestration
// ---------------------------------------------------------------------------
extern "C" void kernel_launch(void* const* d_in, const int* in_sizes, int n_in,
                              void* d_out, int out_size, void* d_ws, size_t ws_size,
                              hipStream_t stream) {
    (void)in_sizes; (void)n_in; (void)out_size; (void)ws_size;
    const int*   seqs  = (const int*)  d_in[0];
    const float* cords = (const float*)d_in[1];
    const float* mask  = (const float*)d_in[2];
    const float* emb   = (const float*)d_in[3];
    float* out = (float*)d_out;

    float* cur = (float*)d_ws;
    // f32 scratch
    float* qb   = cur; cur += (size_t)NROWS*D2F;
    float* kb   = cur; cur += (size_t)NROWS*D2F;
    float* vb   = cur; cur += (size_t)NROWS*D2F;
    float* ab   = cur; cur += (size_t)NROWS*D2F;
    float* Ub   = cur; cur += (size_t)NROWS*1024;
    float* h2   = cur; cur += (size_t)NROWS*HDIM;
    // bf16 scratch (allocated in float units, reinterpreted)
    __bf16* hbufb = (__bf16*)cur; cur += (size_t)NROWS*K0PAD/2;
    __bf16* qbb   = (__bf16*)cur; cur += (size_t)NROWS*D2F/2;
    __bf16* kbb   = (__bf16*)cur; cur += (size_t)NROWS*D2F/2;
    __bf16* abb   = (__bf16*)cur; cur += (size_t)NROWS*D2F/2;
    __bf16* vpk   = (__bf16*)cur; cur += (size_t)BCOLS*16*8*512/2;
    __bf16* h2b   = (__bf16*)cur; cur += (size_t)NROWS*HDIM/2;
    __bf16* wqpk  = (__bf16*)cur; cur += 9*8*512/2;
    __bf16* wkpk  = (__bf16*)cur; cur += 4*8*512/2;
    __bf16* wvpk  = (__bf16*)cur; cur += 4*8*512/2;
    __bf16* wupk  = (__bf16*)cur; cur += 4*64*512/2;
    float* Vmat = cur; cur += 72;
    float* cen  = cur; cur += 24;
    float* padf = cur; cur += 8*LSEQ;

    // stage 0
    eigh_kernel<<<1, 32, 0, stream>>>(cords, mask, Vmat, cen);
    pad_kernel<<<(8*LSEQ + 255)/256, 256, 0, stream>>>(mask, padf);
    {
        size_t total = (size_t)NROWS * K0PAD;
        build_h0<<<(unsigned)((total + 255)/256), 256, 0, stream>>>(
            seqs, cords, mask, emb, Vmat, cen, hbufb);
    }

    for (int layer = 0; layer < 2; ++layer) {
        const float* Wq = (const float*)d_in[4 + 8*layer + 0];
        const float* Wk = (const float*)d_in[4 + 8*layer + 1];
        const float* Wv = (const float*)d_in[4 + 8*layer + 2];
        const float* Wu = (const float*)d_in[4 + 8*layer + 3];
        const float* vf = (const float*)d_in[4 + 8*layer + 4];
        const float* vr = (const float*)d_in[4 + 8*layer + 5];
        const float* bfp = (const float*)d_in[4 + 8*layer + 6];
        const float* brp = (const float*)d_in[4 + 8*layer + 7];

        const __bf16* Ain = layer ? h2b : hbufb;
        const int lda  = layer ? HDIM : K0PAD;
        const int Kpad = layer ? HDIM : K0PAD;
        const int Kact = layer ? HDIM : 259;
        const int Kc   = Kpad / 32;

        // pack weights into fragment order (zero-padded)
        pack_weight<<<(Kc*8*512 + 255)/256, 256, 0, stream>>>(Wq, wqpk, Kact, Kc, 8, D2F);
        pack_weight<<<(4*8*512 + 255)/256, 256, 0, stream>>>(Wk, wkpk, D2F, 4, 8, D2F);
        pack_weight<<<(4*8*512 + 255)/256, 256, 0, stream>>>(Wv, wvpk, D2F, 4, 8, D2F);
        pack_weight<<<(4*64*512 + 255)/256, 256, 0, stream>>>(Wu, wupk, D2F, 4, 64, 1024);

        // q = x @ Wq
        gemm_bf16<<<dim3(NROWS/16, 2), 32, 0, stream>>>(Ain, lda, Kpad, wqpk, 8, qb, D2F, qbb);
        // k = q @ Wk, v = q @ Wv
        gemm_bf16<<<dim3(NROWS/16, 2), 32, 0, stream>>>(qbb, D2F, D2F, wkpk, 8, kb, D2F, kbb);
        gemm_bf16<<<dim3(NROWS/16, 2), 32, 0, stream>>>(qbb, D2F, D2F, wvpk, 8, vb, D2F, (__bf16*)nullptr);
        // pack v into fragment order for the attention PV product
        pack_v<<<(BCOLS*16*8*512)/256, 256, 0, stream>>>(vb, vpk);
        // a = softmax(q k^T) v + q
        attn_kernel<<<dim3(LSEQ/16, BCOLS), 32, 0, stream>>>(qbb, kbb, vpk, qb, padf, ab, abb);
        // U = a @ Wu
        gemm_bf16<<<dim3(NROWS/16, 16), 32, 0, stream>>>(abb, D2F, D2F, wupk, 64, Ub, 1024, (__bf16*)nullptr);
        // bidirectional SRU scan -> h2 (f32) + h2b (bf16)
        sru_scan<<<(BCOLS*2*D2F)/256, 256, 0, stream>>>(Ub, padf, vf, vr, bfp, brp, h2, h2b);
    }

    finalize_mean<<<(8*LSEQ*HDIM)/256, 256, 0, stream>>>(h2, out);
}
